// TriangleAttention_32641751450095
// MI455X (gfx1250) — matrix operand order repaired and checked by reading, not compiled, not run on hardware
//
#include <hip/hip_runtime.h>
#include <hip/hip_bf16.h>
#include <math.h>
#include <stdint.h>

typedef __attribute__((ext_vector_type(16))) _Float16 v16h;
typedef __attribute__((ext_vector_type(8)))  float    v8f;
typedef _Float16 h16;

#define N_DIM 384
#define C_DIM 128
#define H_HEADS 4
#define CH 32
#define NPIX (N_DIM * N_DIM)   // 147456

#if __has_builtin(__builtin_amdgcn_global_load_async_to_lds_b128) && \
    __has_builtin(__builtin_amdgcn_s_wait_asynccnt)
#define HAVE_ASYNC_LDS 1
#else
#define HAVE_ASYNC_LDS 0
#endif

// exact pointee type the async-LDS builtins expect: int4 (gcc vector_size)
typedef int v4i_t __attribute__((vector_size(16)));
typedef __attribute__((address_space(1))) v4i_t gbl_v4i;
typedef __attribute__((address_space(3))) v4i_t lds_v4i;

__device__ __forceinline__ v8f wmma_f16(v16h a, v16h b, v8f c) {
  return __builtin_amdgcn_wmma_f32_16x16x32_f16(false, a, false, b, (short)0, c, false, false);
}

__device__ __forceinline__ float fast_rcp(float x) {
  return __builtin_amdgcn_rcpf(x);   // v_rcp_f32
}

// ---------------------------------------------------------------------------
// Kernel 1: transpose + f32->f16 convert the five 128x128 weight matrices.
// dst[o*128 + i] = src[i*128 + o]  (so B-operand loads are contiguous per lane)
// ---------------------------------------------------------------------------
__global__ __launch_bounds__(256) void wconv_kernel(
    const float* __restrict__ wq, const float* __restrict__ wk,
    const float* __restrict__ wv, const float* __restrict__ wg,
    const float* __restrict__ wo, h16* __restrict__ wT) {
  int mid = blockIdx.y;
  const float* src = (mid == 0) ? wq : (mid == 1) ? wk : (mid == 2) ? wv
                    : (mid == 3) ? wg : wo;
  h16* dst = wT + (size_t)mid * C_DIM * C_DIM;
  int t  = blockIdx.x * 256 + threadIdx.x;       // 0..16383
  int o  = t >> 7;
  int i2 = t & 127;
  dst[o * C_DIM + i2] = (h16)src[i2 * C_DIM + o];
}

// ---------------------------------------------------------------------------
// Kernel 2: fused LayerNorm (wave per pixel) + tri = xn @ w_tri.
// Writes xn as f16 [p][128] and tri f32 [p][4].
// ---------------------------------------------------------------------------
__global__ __launch_bounds__(256) void ln_tri_kernel(
    const float* __restrict__ x, const float* __restrict__ ln_g,
    const float* __restrict__ ln_b, const float* __restrict__ w_tri,
    h16* __restrict__ xn, float* __restrict__ tri) {
  int wave = threadIdx.x >> 5;
  int lane = threadIdx.x & 31;
  int p = blockIdx.x * 8 + wave;
  int c0 = lane * 4;
  const float4 xv = *(const float4*)(x + (size_t)p * C_DIM + c0);

  float s = xv.x + xv.y + xv.z + xv.w;
#pragma unroll
  for (int m = 1; m < 32; m <<= 1) s += __shfl_xor(s, m, 32);
  float mu = s * (1.0f / 128.0f);

  float dx = xv.x - mu, dy = xv.y - mu, dz = xv.z - mu, dw = xv.w - mu;
  float v2 = dx * dx + dy * dy + dz * dz + dw * dw;
#pragma unroll
  for (int m = 1; m < 32; m <<= 1) v2 += __shfl_xor(v2, m, 32);
  float rstd = rsqrtf(v2 * (1.0f / 128.0f) + 1e-5f);

  const float4 gv = *(const float4*)(ln_g + c0);
  const float4 bv = *(const float4*)(ln_b + c0);
  float n0 = dx * rstd * gv.x + bv.x;
  float n1 = dy * rstd * gv.y + bv.y;
  float n2 = dz * rstd * gv.z + bv.z;
  float n3 = dw * rstd * gv.w + bv.w;

  h16* xr = xn + (size_t)p * C_DIM + c0;
  xr[0] = (h16)n0; xr[1] = (h16)n1; xr[2] = (h16)n2; xr[3] = (h16)n3;

  // tri bias: 4 dot products over 128 channels
  float t0 = n0 * w_tri[(c0 + 0) * 4 + 0] + n1 * w_tri[(c0 + 1) * 4 + 0] +
             n2 * w_tri[(c0 + 2) * 4 + 0] + n3 * w_tri[(c0 + 3) * 4 + 0];
  float t1 = n0 * w_tri[(c0 + 0) * 4 + 1] + n1 * w_tri[(c0 + 1) * 4 + 1] +
             n2 * w_tri[(c0 + 2) * 4 + 1] + n3 * w_tri[(c0 + 3) * 4 + 1];
  float t2 = n0 * w_tri[(c0 + 0) * 4 + 2] + n1 * w_tri[(c0 + 1) * 4 + 2] +
             n2 * w_tri[(c0 + 2) * 4 + 2] + n3 * w_tri[(c0 + 3) * 4 + 2];
  float t3 = n0 * w_tri[(c0 + 0) * 4 + 3] + n1 * w_tri[(c0 + 1) * 4 + 3] +
             n2 * w_tri[(c0 + 2) * 4 + 3] + n3 * w_tri[(c0 + 3) * 4 + 3];
#pragma unroll
  for (int m = 1; m < 32; m <<= 1) {
    t0 += __shfl_xor(t0, m, 32);
    t1 += __shfl_xor(t1, m, 32);
    t2 += __shfl_xor(t2, m, 32);
    t3 += __shfl_xor(t3, m, 32);
  }
  if (lane == 0) {
    float4 tv = make_float4(t0, t1, t2, t3);
    *(float4*)(tri + (size_t)p * 4) = tv;
  }
}

// ---------------------------------------------------------------------------
// Kernel 3: WMMA projections q,k,v,g = xn @ W.  One wave computes a 16x128
// output tile. widx = blockIdx.y selects the weight / output layout:
//   0: q -> [i][h][j][32]   1: k -> [i][h][j][32]
//   2: v -> [i][h][32][j] (transposed for the AV B-operand)
//   3: g -> sigmoid(. + bg) -> [p][128]
// ---------------------------------------------------------------------------
__global__ __launch_bounds__(256) void proj_kernel(
    const h16* __restrict__ xn, const h16* __restrict__ wT,
    const float* __restrict__ bg, h16* __restrict__ qo, h16* __restrict__ ko,
    h16* __restrict__ vo, h16* __restrict__ go) {
  int wave  = threadIdx.x >> 5;
  int lane  = threadIdx.x & 31;
  int ptile = blockIdx.x * 8 + wave;
  int widx  = blockIdx.y;
  const h16* w = wT + (size_t)widx * C_DIM * C_DIM;

  int m    = lane & 15;
  int kb   = (lane < 16) ? 0 : 8;    // unused (A also uses cb)
  int cb   = (lane < 16) ? 0 : 16;   // B-matrix K half base
  int n    = lane & 15;              // column (N)
  int Moff = (lane >= 16) ? 8 : 0;   // C/D row base

  const h16* arow = xn + (size_t)(ptile * 16 + m) * C_DIM;
  __builtin_prefetch(arow + 128 * C_DIM, 0, 1);  // next 8-wave block's row
  v16h A[4];
#pragma unroll
  for (int kc = 0; kc < 4; ++kc) {
#pragma unroll
    for (int t = 0; t < 8; ++t) {
      A[kc][t]     = arow[kc * 32 + cb + t];
      A[kc][8 + t] = arow[kc * 32 + cb + 8 + t];
    }
  }

  for (int nb = 0; nb < 8; ++nb) {
    int ncol = nb * 16 + n;
    v8f acc = {};
#pragma unroll
    for (int kc = 0; kc < 4; ++kc) {
      v16h B;
#pragma unroll
      for (int t = 0; t < 16; ++t) B[t] = w[(size_t)ncol * C_DIM + kc * 32 + cb + t];
      acc = wmma_f16(A[kc], B, acc);
    }
#pragma unroll
    for (int r = 0; r < 8; ++r) {
      int p  = ptile * 16 + Moff + r;
      int i  = p / N_DIM;
      int j  = p - i * N_DIM;
      int h  = ncol >> 5;
      int ch = ncol & 31;
      float val = acc[r];
      if (widx == 0) {
        qo[(((size_t)i * H_HEADS + h) * N_DIM + j) * CH + ch] = (h16)val;
      } else if (widx == 1) {
        ko[(((size_t)i * H_HEADS + h) * N_DIM + j) * CH + ch] = (h16)val;
      } else if (widx == 2) {
        vo[(((size_t)i * H_HEADS + h) * CH + ch) * N_DIM + j] = (h16)val;
      } else {
        // sigmoid with v_rcp_f32 instead of IEEE divide
        float gv = fast_rcp(1.0f + __expf(-(val + bg[ncol])));
        go[(size_t)p * C_DIM + ncol] = (h16)gv;
      }
    }
  }
}

// ---------------------------------------------------------------------------
// Kernel 4: flash attention per (i, h). Block = 4 waves x 16 queries = 64 q.
// k and v^T for the (i,h) row staged in LDS (async global->LDS DMA when the
// toolchain exposes it, tracked by ASYNCcnt); online softmax over 12 chunks
// of 32 keys; gate + normalize in the epilogue.
// ---------------------------------------------------------------------------
__global__ __launch_bounds__(128) void attn_kernel(
    const h16* __restrict__ q, const h16* __restrict__ k,
    const h16* __restrict__ vT, const float* __restrict__ tri,
    const float* __restrict__ mask, const h16* __restrict__ g,
    h16* __restrict__ o) {
  __shared__ h16 ksh[N_DIM * CH];     // [key][ch]   24 KB
  __shared__ h16 vsh[CH * N_DIM];     // [ch][key]   24 KB
  __shared__ h16 psh[4][16 * 32];     // per-wave P scratch, 4 KB

  int i = blockIdx.z, h = blockIdx.y;
  int wave = threadIdx.x >> 5;
  int lane = threadIdx.x & 31;

  size_t base = ((size_t)i * H_HEADS + h) * (N_DIM * CH);
#if HAVE_ASYNC_LDS
  { // async DMA global -> LDS, tracked by ASYNCcnt (no VGPR round-trip)
    const char* kg = (const char*)(k + base);
    const char* vg = (const char*)(vT + base);
    char* ks = (char*)ksh;
    char* vs = (char*)vsh;
    for (int t = threadIdx.x; t < (N_DIM * CH * 2) / 16; t += 128) {
      __builtin_amdgcn_global_load_async_to_lds_b128(
          (gbl_v4i*)(uintptr_t)(kg + t * 16),
          (lds_v4i*)(uintptr_t)(ks + t * 16), 0, 0);
      __builtin_amdgcn_global_load_async_to_lds_b128(
          (gbl_v4i*)(uintptr_t)(vg + t * 16),
          (lds_v4i*)(uintptr_t)(vs + t * 16), 0, 0);
    }
    __builtin_amdgcn_s_wait_asynccnt(0);
  }
#else
  { // cooperative stage of k and v^T (6144 dwords each)
    const unsigned int* kg = (const unsigned int*)(k + base);
    const unsigned int* vg = (const unsigned int*)(vT + base);
    unsigned int* ks4 = (unsigned int*)ksh;
    unsigned int* vs4 = (unsigned int*)vsh;
    for (int t = threadIdx.x; t < (N_DIM * CH) / 2; t += 128) {
      ks4[t] = kg[t];
      vs4[t] = vg[t];
    }
  }
#endif
  __syncthreads();

  int qbase = blockIdx.x * 64 + wave * 16;
  int m    = lane & 15;
  int kb   = (lane < 16) ? 0 : 8;
  int cb   = (lane < 16) ? 0 : 16;
  int n    = lane & 15;
  int Moff = (lane >= 16) ? 8 : 0;

  // q tile in A layout
  const h16* qrow = q + base + (size_t)(qbase + m) * CH;
  v16h aq;
#pragma unroll
  for (int t = 0; t < 8; ++t) { aq[t] = qrow[cb + t]; aq[8 + t] = qrow[cb + 8 + t]; }

  v8f oa0 = {}, oa1 = {};
  float rmax[8], rsum[8];
#pragma unroll
  for (int r = 0; r < 8; ++r) { rmax[r] = -INFINITY; rsum[r] = 0.0f; }

  const float scale = 0.17677669529663687f;  // 1/sqrt(32)

  for (int kt = 0; kt < 12; ++kt) {
    int jb = kt * 32;
    // score tiles: A = q (16x32 ch), B = k^T (32ch x 16 keys)
    v16h b0, b1;
#pragma unroll
    for (int t = 0; t < 16; ++t) {
      b0[t] = ksh[(jb + n) * CH + cb + t];
      b1[t] = ksh[(jb + 16 + n) * CH + cb + t];
    }
    v8f zc = {};
    v8f s0 = wmma_f16(aq, b0, zc);
    v8f s1 = wmma_f16(aq, b1, zc);

    float mb0 = 1e9f * (mask[(size_t)i * N_DIM + jb + n] - 1.0f);
    float mb1 = 1e9f * (mask[(size_t)i * N_DIM + jb + 16 + n] - 1.0f);
#pragma unroll
    for (int r = 0; r < 8; ++r) {
      int qr = qbase + Moff + r;
      s0[r] = s0[r] * scale + mb0 + tri[((size_t)qr * N_DIM + jb + n) * 4 + h];
      s1[r] = s1[r] * scale + mb1 + tri[((size_t)qr * N_DIM + jb + 16 + n) * 4 + h];
    }
    // online softmax per row (rows live across the 16-lane halves)
#pragma unroll
    for (int r = 0; r < 8; ++r) {
      float mt = fmaxf(s0[r], s1[r]);
#pragma unroll
      for (int mm = 1; mm < 16; mm <<= 1) mt = fmaxf(mt, __shfl_xor(mt, mm, 32));
      float mnew = fmaxf(rmax[r], mt);
      float corr = __expf(rmax[r] - mnew);
      float p0 = __expf(s0[r] - mnew);
      float p1 = __expf(s1[r] - mnew);
      float rs = p0 + p1;
#pragma unroll
      for (int mm = 1; mm < 16; mm <<= 1) rs += __shfl_xor(rs, mm, 32);
      rsum[r] = rsum[r] * corr + rs;
      rmax[r] = mnew;
      oa0[r] *= corr;
      oa1[r] *= corr;
      // D-layout -> row-major f16 in LDS for A-layout reload
      psh[wave][(Moff + r) * 32 + n]      = (h16)p0;
      psh[wave][(Moff + r) * 32 + 16 + n] = (h16)p1;
    }
    // P in A layout (16 q x 32 keys)
    v16h pa;
#pragma unroll
    for (int t = 0; t < 8; ++t) {
      pa[t]     = psh[wave][m * 32 + cb + t];
      pa[8 + t] = psh[wave][m * 32 + cb + 8 + t];
    }
    // V chunk as B (32 keys x 16 ch), two ch tiles, from transposed v
    v16h bv0, bv1;
#pragma unroll
    for (int t = 0; t < 16; ++t) {
      bv0[t] = vsh[(size_t)n * N_DIM + jb + cb + t];
      bv1[t] = vsh[(size_t)(16 + n) * N_DIM + jb + cb + t];
    }
    oa0 = wmma_f16(pa, bv0, oa0);
    oa1 = wmma_f16(pa, bv1, oa1);
  }

  // epilogue: normalize, gate, store o as f16 [p][128]
#pragma unroll
  for (int r = 0; r < 8; ++r) {
    float inv = fast_rcp(rsum[r]);
    int qr = qbase + Moff + r;
    size_t gb = ((size_t)i * N_DIM + qr) * C_DIM + (size_t)h * CH;
    float g0 = (float)g[gb + n];
    float g1 = (float)g[gb + 16 + n];
    o[gb + n]      = (h16)(oa0[r] * inv * g0);
    o[gb + 16 + n] = (h16)(oa1[r] * inv * g1);
  }
}

// ---------------------------------------------------------------------------
// Kernel 5: out = o_gated @ wo + bo, f32 output.
// ---------------------------------------------------------------------------
__global__ __launch_bounds__(256) void outproj_kernel(
    const h16* __restrict__ oin, const h16* __restrict__ woT,
    const float* __restrict__ bo, float* __restrict__ out) {
  int wave  = threadIdx.x >> 5;
  int lane  = threadIdx.x & 31;
  int ptile = blockIdx.x * 8 + wave;

  int m    = lane & 15;
  int kb   = (lane < 16) ? 0 : 8;
  int cb   = (lane < 16) ? 0 : 16;
  int n    = lane & 15;
  int Moff = (lane >= 16) ? 8 : 0;

  const h16* arow = oin + (size_t)(ptile * 16 + m) * C_DIM;
  __builtin_prefetch(arow + 128 * C_DIM, 0, 1);
  v16h A[4];
#pragma unroll
  for (int kc = 0; kc < 4; ++kc) {
#pragma unroll
    for (int t = 0; t < 8; ++t) {
      A[kc][t]     = arow[kc * 32 + cb + t];
      A[kc][8 + t] = arow[kc * 32 + cb + 8 + t];
    }
  }

  for (int nb = 0; nb < 8; ++nb) {
    int ncol = nb * 16 + n;
    v8f acc = {};
#pragma unroll
    for (int kc = 0; kc < 4; ++kc) {
      v16h B;
#pragma unroll
      for (int t = 0; t < 16; ++t)
        B[t] = woT[(size_t)ncol * C_DIM + kc * 32 + cb + t];
      acc = wmma_f16(A[kc], B, acc);
    }
    float bias = bo[ncol];
#pragma unroll
    for (int r = 0; r < 8; ++r) {
      int p = ptile * 16 + Moff + r;
      out[(size_t)p * C_DIM + ncol] = acc[r] + bias;
    }
  }
}

// ---------------------------------------------------------------------------
extern "C" void kernel_launch(void* const* d_in, const int* in_sizes, int n_in,
                              void* d_out, int out_size, void* d_ws,
                              size_t ws_size, hipStream_t stream) {
  const float* x     = (const float*)d_in[0];
  const float* mask  = (const float*)d_in[1];
  const float* ln_g  = (const float*)d_in[2];
  const float* ln_b  = (const float*)d_in[3];
  const float* w_tri = (const float*)d_in[4];
  const float* wq    = (const float*)d_in[5];
  const float* wk    = (const float*)d_in[6];
  const float* wv    = (const float*)d_in[7];
  const float* wg    = (const float*)d_in[8];
  const float* bg    = (const float*)d_in[9];
  const float* wo    = (const float*)d_in[10];
  const float* bo    = (const float*)d_in[11];
  float* out = (float*)d_out;

  char* ws = (char*)d_ws;
  const size_t SZ = (size_t)NPIX * C_DIM * sizeof(h16);  // 37,748,736 B
  h16*   xn  = (h16*)(ws + 0 * SZ);
  h16*   qb  = (h16*)(ws + 1 * SZ);
  h16*   kb  = (h16*)(ws + 2 * SZ);
  h16*   vb  = (h16*)(ws + 3 * SZ);
  h16*   gb  = (h16*)(ws + 4 * SZ);
  h16*   ob  = (h16*)(ws + 5 * SZ);
  float* tri = (float*)(ws + 6 * SZ);
  h16*   wT  = (h16*)(ws + 6 * SZ + (size_t)NPIX * 4 * sizeof(float));

  // 1) weights -> f16, transposed
  hipLaunchKernelGGL(wconv_kernel, dim3(64, 5), dim3(256), 0, stream,
                     wq, wk, wv, wg, wo, wT);
  // 2) layernorm + tri bias
  hipLaunchKernelGGL(ln_tri_kernel, dim3(NPIX / 8), dim3(256), 0, stream,
                     x, ln_g, ln_b, w_tri, xn, tri);
  // 3) q,k,v,g projections (WMMA)
  hipLaunchKernelGGL(proj_kernel, dim3(NPIX / 16 / 8, 4), dim3(256), 0, stream,
                     xn, wT, bg, qb, kb, vb, gb);
  // 4) flash attention per (i,h) (WMMA + async LDS staging)
  hipLaunchKernelGGL(attn_kernel, dim3(6, H_HEADS, N_DIM), dim3(128), 0, stream,
                     qb, kb, vb, tri, mask, gb, ob);
  // 5) output projection (WMMA)
  hipLaunchKernelGGL(outproj_kernel, dim3(NPIX / 16 / 8), dim3(256), 0, stream,
                     ob, wT + 4 * C_DIM * C_DIM, bo, out);
}
